// MHSAModule_57380763075245
// MI455X (gfx1250) — matrix operand order repaired and checked by reading, not compiled, not run on hardware
//
#include <hip/hip_runtime.h>
#include <hip/hip_bf16.h>

// ---------------------------------------------------------------------------
// Transformer-XL MHSA block for MI455X (gfx1250, wave32, WMMA).
// All matmuls run on v_wmma_f32_16x16x32_bf16 (bf16 operands, f32 accum).
// All WMMA B-operands are pre-packed into the fragment-native lane layout so
// each fragment load is one aligned 32-byte vector load (2 x global_load_b128).
// ---------------------------------------------------------------------------

typedef __bf16 bf16;
typedef __attribute__((ext_vector_type(16))) __bf16 v16bf;
typedef __attribute__((ext_vector_type(8)))  float  v8f;

#define D_MODEL 512
#define T_SEQ   1024
#define N_B     8
#define N_H     8
#define DH      64
#define BT      (N_B * T_SEQ)   // 8192
#define QB      32              // query rows per attention workgroup

__device__ __forceinline__ v8f wmma_bf16(v16bf a, v16bf b, v8f c) {
  // D = A(16x32) * B(32x16) + C, f32 accumulate
  return __builtin_amdgcn_wmma_f32_16x16x32_bf16(
      /*neg_a=*/false, a, /*neg_b=*/false, b,
      /*c_mod=*/(short)0, c, /*reuse_a=*/false, /*reuse_b=*/false);
}

// ---- fragment helpers (layouts per CDNA5 ISA section 7.12.2) ---------------
// A 16x32 bf16 (MxK), row-major source with leading dim ld:
//   lane = hh*16 + m ; VGPR v<4: K = hh*8 + 2v + w ; v>=4: K = 16 + hh*8 + 2(v-4) + w
// -> per lane this is two contiguous 8-element (16B) runs; compiler merges to b128s.
__device__ __forceinline__ v16bf load_a_g(const bf16* base, int ld) {
  int lane = threadIdx.x & 31, m = lane & 15, hh = lane >> 4;
  v16bf f;
#pragma unroll
  for (int e = 0; e < 16; ++e) {
    int v = e >> 1, w = e & 1;
    int k = (v < 4) ? (hh * 8 + v * 2 + w) : (16 + hh * 8 + (v - 4) * 2 + w);
    f[e] = base[(size_t)m * ld + k];
  }
  return f;
}
// Packed B tile: 32x16 fragment stored as [lane][16 bf16] (1 KB per tile).
__device__ __forceinline__ v16bf load_b_packed(const bf16* tileBase) {
  int lane = threadIdx.x & 31;
  return reinterpret_cast<const v16bf*>(tileBase)[lane];  // 32B aligned vector load
}
// A 16x32 fragment gathered from f32 LDS with on-the-fly bf16 convert.
__device__ __forceinline__ v16bf load_a_lds_f32(const float* base, int ld) {
  int lane = threadIdx.x & 31, m = lane & 15, hh = lane >> 4;
  v16bf f;
#pragma unroll
  for (int e = 0; e < 16; ++e) {
    int v = e >> 1, w = e & 1;
    int k = (v < 4) ? (hh * 8 + v * 2 + w) : (16 + hh * 8 + (v - 4) * 2 + w);
    f[e] = (bf16)base[m * ld + k];
  }
  return f;
}
// B-fragment element coordinates for pack kernels: lane r in [0,32), e in [0,16)
// maps to (k = v<4 ? hh*8 + 2v + w : 16 + hh*8 + 2(v-4) + w, n = lm).
__device__ __forceinline__ void b_coords(int lane, int e, int& k, int& n) {
  int lm = lane & 15, hh = lane >> 4;
  int v = e >> 1, w = e & 1;
  k = (v < 4) ? (hh * 8 + v * 2 + w) : (16 + hh * 8 + (v - 4) * 2 + w);
  n = lm;
}

// ---------------------------------------------------------------------------
// Kernel 1: pre-LayerNorm, f32 in -> bf16 out. One block per (b,t) row.
// ---------------------------------------------------------------------------
__global__ __launch_bounds__(256) void ln_kernel(const float* __restrict__ x,
                                                 const float* __restrict__ g,
                                                 const float* __restrict__ be,
                                                 bf16* __restrict__ xn) {
  __shared__ float red[256];
  int row = blockIdx.x, t = threadIdx.x;
  const float* xr = x + (size_t)row * D_MODEL;
  float v0 = xr[t], v1 = xr[t + 256];
  red[t] = v0 + v1;
  __syncthreads();
  for (int s = 128; s > 0; s >>= 1) { if (t < s) red[t] += red[t + s]; __syncthreads(); }
  float mu = red[0] * (1.0f / D_MODEL);
  __syncthreads();
  float d0 = v0 - mu, d1 = v1 - mu;
  red[t] = d0 * d0 + d1 * d1;
  __syncthreads();
  for (int s = 128; s > 0; s >>= 1) { if (t < s) red[t] += red[t + s]; __syncthreads(); }
  float rstd = rsqrtf(red[0] * (1.0f / D_MODEL) + 1e-5f);
  bf16* o = xn + (size_t)row * D_MODEL;
  o[t]       = (bf16)(d0 * rstd * g[t] + be[t]);
  o[t + 256] = (bf16)(d1 * rstd * g[t + 256] + be[t + 256]);
}

// ---------------------------------------------------------------------------
// Kernel 2: sinusoidal PE (TXL order, pos = T-1..0) -> bf16. One block per row.
// ---------------------------------------------------------------------------
__global__ __launch_bounds__(256) void pe_kernel(bf16* __restrict__ pe) {
  int j = blockIdx.x, t = threadIdx.x;
  float pos = (float)(T_SEQ - 1 - j);
  float inv = __expf(-((float)(2 * t) / (float)D_MODEL) * 9.210340371976184f);
  float ang = pos * inv;
  pe[(size_t)j * D_MODEL + t]       = (bf16)__sinf(ang);
  pe[(size_t)j * D_MODEL + t + 256] = (bf16)__cosf(ang);
}

// ---------------------------------------------------------------------------
// Pack kernels: build fragment-native B-operand tiles (each thread produces
// one aligned 16-byte store = 8 packed bf16).
// Tile order for a KxN operand: idx = nt*(K/32) + kt ; tile size 512 elems.
// ---------------------------------------------------------------------------
// (a) weights: f32 row-major KxN
__global__ __launch_bounds__(256) void pack_w(const float* __restrict__ src,
                                              bf16* __restrict__ dst, int K, int N) {
  int gid = blockIdx.x * 256 + threadIdx.x;   // K*N/8 groups
  int tile = gid >> 6, r = gid & 63;
  int lane = r >> 1, h2 = r & 1;
  int ktiles = K >> 5;
  int nt = tile / ktiles, kt = tile - nt * ktiles;
  alignas(16) bf16 tmp[8];
#pragma unroll
  for (int j = 0; j < 8; ++j) {
    int k, n;
    b_coords(lane, h2 * 8 + j, k, n);
    tmp[j] = (bf16)src[(size_t)(kt * 32 + k) * N + nt * 16 + n];
  }
  *(uint4*)(dst + (size_t)tile * 512 + lane * 16 + h2 * 8) = *(const uint4*)tmp;
}
// (b) K^T per (b,h): B[kdim][t] = qkv[(b*T+t)*1536 + 512 + h*64 + kdim]; K=64,N=1024
__global__ __launch_bounds__(256) void pack_kT(const float* __restrict__ qkv,
                                               bf16* __restrict__ dst) {
  int h = blockIdx.y, b = blockIdx.z;
  int gid = blockIdx.x * 256 + threadIdx.x;   // 8192 groups per (b,h)
  int tile = gid >> 6, r = gid & 63;
  int lane = r >> 1, h2 = r & 1;
  int nt = tile >> 1, kt = tile & 1;
  alignas(16) bf16 tmp[8];
#pragma unroll
  for (int j = 0; j < 8; ++j) {
    int k, n;
    b_coords(lane, h2 * 8 + j, k, n);
    int t = nt * 16 + n, kd = kt * 32 + k;
    tmp[j] = (bf16)qkv[(size_t)(b * T_SEQ + t) * (3 * D_MODEL) + D_MODEL + h * DH + kd];
  }
  *(uint4*)(dst + ((size_t)(b * N_H + h)) * 65536 + (size_t)tile * 512 + lane * 16 + h2 * 8) =
      *(const uint4*)tmp;
}
// (c) R^T per h: B[kdim][j] = rf[j*512 + h*64 + kdim]; K=64, N=1024
__global__ __launch_bounds__(256) void pack_rT(const float* __restrict__ rf,
                                               bf16* __restrict__ dst) {
  int h = blockIdx.y;
  int gid = blockIdx.x * 256 + threadIdx.x;
  int tile = gid >> 6, r = gid & 63;
  int lane = r >> 1, h2 = r & 1;
  int nt = tile >> 1, kt = tile & 1;
  alignas(16) bf16 tmp[8];
#pragma unroll
  for (int j = 0; j < 8; ++j) {
    int k, n;
    b_coords(lane, h2 * 8 + j, k, n);
    tmp[j] = (bf16)rf[(size_t)(nt * 16 + n) * D_MODEL + h * DH + kt * 32 + k];
  }
  *(uint4*)(dst + (size_t)h * 65536 + (size_t)tile * 512 + lane * 16 + h2 * 8) =
      *(const uint4*)tmp;
}
// (d) V per (b,h): B[kk][dh] over 32-key chunks; tile idx = tt*4 + nt (tt=t/32)
__global__ __launch_bounds__(256) void pack_v(const float* __restrict__ qkv,
                                              bf16* __restrict__ dst) {
  int h = blockIdx.y, b = blockIdx.z;
  int gid = blockIdx.x * 256 + threadIdx.x;   // 8192 groups per (b,h)
  int tile = gid >> 6, r = gid & 63;
  int lane = r >> 1, h2 = r & 1;
  int tt = tile >> 2, nt = tile & 3;
  alignas(16) bf16 tmp[8];
#pragma unroll
  for (int j = 0; j < 8; ++j) {
    int k, n;
    b_coords(lane, h2 * 8 + j, k, n);
    int t = tt * 32 + k;
    tmp[j] = (bf16)qkv[(size_t)(b * T_SEQ + t) * (3 * D_MODEL) + 2 * D_MODEL + h * DH +
                       nt * 16 + n];
  }
  *(uint4*)(dst + ((size_t)(b * N_H + h)) * 65536 + (size_t)tile * 512 + lane * 16 + h2 * 8) =
      *(const uint4*)tmp;
}

// ---------------------------------------------------------------------------
// Kernel: generic WMMA GEMM, packed-B operand. Block = 8 waves arranged
// 4(m) x 2(n); wave tile 16x64 (one A fragment feeds 4 WMMAs per k-step).
// Block tile 64x128. M % 64 == 0, N % 128 == 0, K % 32 == 0.
// ---------------------------------------------------------------------------
__global__ __launch_bounds__(256) void gemm_bf16_kernel(
    const bf16* __restrict__ A, const bf16* __restrict__ Bp,
    float* __restrict__ C, const float* __restrict__ resid,
    bf16* __restrict__ Cbf, int K, int lda, int ldc) {
  int w = threadIdx.x >> 5, lane = threadIdx.x & 31;
  int lm = lane & 15, hh = lane >> 4;
  int mt = w & 3, ntw = w >> 2;
  int m0 = blockIdx.y * 64 + mt * 16;
  int nt0 = (blockIdx.x * 128 + ntw * 64) >> 4;   // base n-tile index
  int ktiles = K >> 5;
  v8f acc[4] = {{}, {}, {}, {}};
  const bf16* aRow = A + (size_t)m0 * lda;
  for (int kt = 0; kt < ktiles; ++kt) {
    v16bf a = load_a_g(aRow + kt * 32, lda);
    if (kt + 1 < ktiles)
      __builtin_prefetch(aRow + (kt + 1) * 32, 0, 3);   // global_prefetch_b8
#pragma unroll
    for (int nt = 0; nt < 4; ++nt) {
      v16bf b = load_b_packed(Bp + ((size_t)(nt0 + nt) * ktiles + kt) * 512);
      acc[nt] = wmma_bf16(a, b, acc[nt]);
    }
  }
#pragma unroll
  for (int nt = 0; nt < 4; ++nt) {
#pragma unroll
    for (int j = 0; j < 8; ++j) {
      int m = m0 + 8 * hh + j;
      size_t i0 = (size_t)m * ldc + (nt0 + nt) * 16 + lm;
      float v0 = acc[nt][j];
      if (resid) v0 += resid[i0];
      if (C)   C[i0] = v0;
      if (Cbf) Cbf[i0] = (bf16)v0;
    }
  }
}

// ---------------------------------------------------------------------------
// Kernel: QKV epilogue for Q only: add r_w_bias / r_r_bias, bf16-ify.
// (K and V go straight from qkv f32 into packed tiles.)
// ---------------------------------------------------------------------------
__global__ __launch_bounds__(256) void qkv_epi(
    const float* __restrict__ qkv, const float* __restrict__ rwb,
    const float* __restrict__ rrb, bf16* __restrict__ qrw, bf16* __restrict__ qrr) {
  int i = blockIdx.x * 256 + threadIdx.x;  // i < BT*512
  int bt = i >> 9, c = i & 511;            // c = h*64 + dh
  float q = qkv[(size_t)bt * (3 * D_MODEL) + c];
  qrw[i] = (bf16)(q + rwb[c]);
  qrr[i] = (bf16)(q + rrb[c]);
}

// ---------------------------------------------------------------------------
// Kernel: attention. One workgroup per (b, h, 32-query block).
// Dynamic LDS: sBD[32][1024] f32 | sS[32][1024] f32 | sWrap[1024] f32 (266 KB).
// rel_shift: score_bd[q][k] = k<=q ? bd[q][T-1-q+k]
//                           : k==q+1 ? 0 : bd[q+1][k-q-2]
// ---------------------------------------------------------------------------
__global__ __launch_bounds__(256) void attn_kernel(
    const bf16* __restrict__ qrw, const bf16* __restrict__ qrr,
    const bf16* __restrict__ kTp, const bf16* __restrict__ vp,
    const bf16* __restrict__ rTp, const float* __restrict__ rf,
    const int* __restrict__ lens, bf16* __restrict__ obf) {
  extern __shared__ char smem_raw[];
  float* sBD   = (float*)smem_raw;       // QB*1024
  float* sS    = sBD + QB * 1024;        // QB*1024
  float* sWrap = sS + QB * 1024;         // 1024

  const int q0 = blockIdx.x * QB;
  const int h  = blockIdx.y;
  const int b  = blockIdx.z;
  const int len = lens[b];
  const int tid = threadIdx.x, w = tid >> 5, lane = tid & 31;
  const int lm = lane & 15, hh = lane >> 4;
  const int qs = w & 1;   // which 16-row query subtile
  const int kr = w >> 1;  // which 256-key slab

  const bf16* qrrA = qrr + ((size_t)(b * T_SEQ + q0 + qs * 16)) * D_MODEL + h * DH;
  const bf16* qrwA = qrw + ((size_t)(b * T_SEQ + q0 + qs * 16)) * D_MODEL + h * DH;
  const bf16* kT = kTp + ((size_t)(b * N_H + h)) * 65536;
  const bf16* vT = vp  + ((size_t)(b * N_H + h)) * 65536;
  const bf16* rT = rTp + (size_t)h * 65536;

  // ---- phase 1: unshifted BD block via WMMA: (q+rrb) @ R^T ----
  {
    v16bf a0 = load_a_g(qrrA, D_MODEL);
    v16bf a1 = load_a_g(qrrA + 32, D_MODEL);
    for (int jt = 0; jt < 16; ++jt) {
      int ntile = kr * 16 + jt;
      v8f acc = {};
      acc = wmma_bf16(a0, load_b_packed(rT + (size_t)(ntile * 2 + 0) * 512), acc);
      acc = wmma_bf16(a1, load_b_packed(rT + (size_t)(ntile * 2 + 1) * 512), acc);
#pragma unroll
      for (int j = 0; j < 8; ++j)
        sBD[(qs * 16 + 8 * hh + j) * 1024 + ntile * 16 + lm] = acc[j];
    }
  }
  // wrap row bd[q0+QB][*] (needed for the above-diagonal rel-shift wrap)
  if (q0 + QB < T_SEQ) {
    const bf16* qx = qrr + ((size_t)(b * T_SEQ + q0 + QB)) * D_MODEL + h * DH;
    for (int j = tid; j < T_SEQ; j += 256) {
      const float* rr = rf + (size_t)j * D_MODEL + h * DH;
      float s = 0.0f;
#pragma unroll 8
      for (int d = 0; d < DH; ++d) s += (float)qx[d] * rr[d];
      sWrap[j] = s;
    }
  }
  __syncthreads();

  // ---- phase 2: AC block via WMMA: (q+rwb) @ K^T ----
  {
    v16bf a0 = load_a_g(qrwA, D_MODEL);
    v16bf a1 = load_a_g(qrwA + 32, D_MODEL);
    for (int jt = 0; jt < 16; ++jt) {
      int ntile = kr * 16 + jt;
      v8f acc = {};
      acc = wmma_bf16(a0, load_b_packed(kT + (size_t)(ntile * 2 + 0) * 512), acc);
      acc = wmma_bf16(a1, load_b_packed(kT + (size_t)(ntile * 2 + 1) * 512), acc);
#pragma unroll
      for (int j = 0; j < 8; ++j)
        sS[(qs * 16 + 8 * hh + j) * 1024 + ntile * 16 + lm] = acc[j];
    }
  }
  __syncthreads();

  // ---- phase 3: combine (AC + shifted BD) * scale, mask padded keys ----
  const float scale = 0.125f;  // 1/sqrt(64)
  for (int i = tid; i < QB * 1024; i += 256) {
    int qrow = i >> 10, k = i & 1023;
    int q = q0 + qrow;
    float v;
    if (k >= len) {
      v = -1e30f;
    } else {
      float bd;
      if (k <= q)           bd = sBD[qrow * 1024 + (T_SEQ - 1 - q + k)];
      else if (k == q + 1)  bd = 0.0f;
      else bd = (qrow + 1 < QB) ? sBD[(qrow + 1) * 1024 + (k - q - 2)]
                                : sWrap[k - q - 2];
      v = (sS[i] + bd) * scale;
    }
    sS[i] = v;
  }
  __syncthreads();

  // ---- phase 4: softmax per row (wave32 shuffle reduction), 4 rows/wave ----
  for (int rr = 0; rr < 4; ++rr) {
    int row = w * 4 + rr;
    float* Sr = sS + row * 1024;
    float mx = -3.0e38f;
    for (int k = lane; k < 1024; k += 32) mx = fmaxf(mx, Sr[k]);
    for (int off = 16; off > 0; off >>= 1) mx = fmaxf(mx, __shfl_xor(mx, off, 32));
    float sum = 0.0f;
    for (int k = lane; k < 1024; k += 32) {
      float e = __expf(Sr[k] - mx);
      Sr[k] = e;
      sum += e;
    }
    for (int off = 16; off > 0; off >>= 1) sum += __shfl_xor(sum, off, 32);
    float invs = 1.0f / sum;
    for (int k = lane; k < 1024; k += 32) Sr[k] *= invs;
  }
  __syncthreads();

  // ---- phase 5: O = P @ V via WMMA (P fragments gathered bf16 from LDS) ----
  v8f o[4] = {{}, {}, {}, {}};
  for (int kc = 0; kc < 8; ++kc) {
    int tt = kr * 8 + kc;                       // 32-key chunk index
    v16bf pa = load_a_lds_f32(sS + (qs * 16) * 1024 + tt * 32, 1024);
#pragma unroll
    for (int nt = 0; nt < 4; ++nt) {
      v16bf bv = load_b_packed(vT + (size_t)(tt * 4 + nt) * 512);
      o[nt] = wmma_bf16(pa, bv, o[nt]);
    }
  }
  // cross-wave partial reduction through LDS (reuse sBD region)
  float* part = sBD;  // [8 slabs][16 rows][64] = 32 KB
#pragma unroll
  for (int nt = 0; nt < 4; ++nt)
#pragma unroll
    for (int j = 0; j < 8; ++j)
      part[((kr * 2 + qs) * 16 + 8 * hh + j) * 64 + nt * 16 + lm] = o[nt][j];
  __syncthreads();
  for (int i = tid; i < 2 * 16 * 64; i += 256) {
    int qss = i >> 10, rem = i & 1023;
    float s = 0.0f;
#pragma unroll
    for (int kk = 0; kk < 4; ++kk) s += part[(kk * 2 + qss) * 1024 + rem];
    int m = rem >> 6, dh = rem & 63;
    obf[((size_t)(b * T_SEQ + q0 + qss * 16 + m)) * D_MODEL + h * DH + dh] = (bf16)s;
  }
}

// ---------------------------------------------------------------------------
__global__ void copy_lens(const int* __restrict__ lens, int* __restrict__ dst) {
  int i = threadIdx.x;
  if (i < N_B) dst[i] = lens[i];
}

// ---------------------------------------------------------------------------
extern "C" void kernel_launch(void* const* d_in, const int* in_sizes, int n_in,
                              void* d_out, int out_size, void* d_ws, size_t ws_size,
                              hipStream_t stream) {
  (void)in_sizes; (void)n_in; (void)out_size; (void)ws_size;
  const float* x     = (const float*)d_in[0];
  const int*   lens  = (const int*)d_in[1];
  const float* gamma = (const float*)d_in[2];
  const float* beta  = (const float*)d_in[3];
  const float* w_qkv = (const float*)d_in[4];
  const float* w_pos = (const float*)d_in[5];
  const float* w_out = (const float*)d_in[6];
  const float* rwb   = (const float*)d_in[7];
  const float* rrb   = (const float*)d_in[8];
  float* out = (float*)d_out;

  // workspace layout (256B aligned slices)
  char* ws = (char*)d_ws;
  size_t off = 0;
  auto alloc = [&](size_t bytes) {
    size_t o = off;
    off += (bytes + 255) & ~(size_t)255;
    return o;
  };
  bf16*  xn     = (bf16*)(ws + alloc((size_t)BT * D_MODEL * 2));
  bf16*  peb    = (bf16*)(ws + alloc((size_t)T_SEQ * D_MODEL * 2));
  bf16*  wqkv_p = (bf16*)(ws + alloc((size_t)D_MODEL * 3 * D_MODEL * 2));
  bf16*  wpos_p = (bf16*)(ws + alloc((size_t)D_MODEL * D_MODEL * 2));
  bf16*  wout_p = (bf16*)(ws + alloc((size_t)D_MODEL * D_MODEL * 2));
  float* qkvf   = (float*)(ws + alloc((size_t)BT * 3 * D_MODEL * 4));
  float* rf     = (float*)(ws + alloc((size_t)T_SEQ * D_MODEL * 4));
  bf16*  qrw    = (bf16*)(ws + alloc((size_t)BT * D_MODEL * 2));
  bf16*  qrr    = (bf16*)(ws + alloc((size_t)BT * D_MODEL * 2));
  bf16*  kTp    = (bf16*)(ws + alloc((size_t)N_B * N_H * 65536 * 2));
  bf16*  vp     = (bf16*)(ws + alloc((size_t)N_B * N_H * 65536 * 2));
  bf16*  rTp    = (bf16*)(ws + alloc((size_t)N_H * 65536 * 2));
  bf16*  attnb  = (bf16*)(ws + alloc((size_t)BT * D_MODEL * 2));

  // 1. LayerNorm -> bf16 ; positional encoding -> bf16
  ln_kernel<<<BT, 256, 0, stream>>>(x, gamma, beta, xn);
  pe_kernel<<<T_SEQ, 256, 0, stream>>>(peb);
  // 2. pack weights into fragment-native B tiles
  pack_w<<<(D_MODEL * 3 * D_MODEL / 8) / 256, 256, 0, stream>>>(w_qkv, wqkv_p,
                                                                D_MODEL, 3 * D_MODEL);
  pack_w<<<(D_MODEL * D_MODEL / 8) / 256, 256, 0, stream>>>(w_pos, wpos_p,
                                                            D_MODEL, D_MODEL);
  pack_w<<<(D_MODEL * D_MODEL / 8) / 256, 256, 0, stream>>>(w_out, wout_p,
                                                            D_MODEL, D_MODEL);
  // 3. QKV projection (8192x512 @ 512x1536 -> f32)
  gemm_bf16_kernel<<<dim3(3 * D_MODEL / 128, BT / 64), 256, 0, stream>>>(
      xn, wqkv_p, qkvf, nullptr, nullptr, D_MODEL, D_MODEL, 3 * D_MODEL);
  // 4. positional projection r = pe @ w_pos (1024x512 @ 512x512 -> f32)
  gemm_bf16_kernel<<<dim3(D_MODEL / 128, T_SEQ / 64), 256, 0, stream>>>(
      peb, wpos_p, rf, nullptr, nullptr, D_MODEL, D_MODEL, D_MODEL);
  // 5. epilogues: q+biases -> bf16 ; pack K^T, V, R^T fragment tiles
  qkv_epi<<<(BT * D_MODEL) / 256, 256, 0, stream>>>(qkvf, rwb, rrb, qrw, qrr);
  pack_kT<<<dim3(32, N_H, N_B), 256, 0, stream>>>(qkvf, kTp);
  pack_v<<<dim3(32, N_H, N_B), 256, 0, stream>>>(qkvf, vp);
  pack_rT<<<dim3(32, N_H), 256, 0, stream>>>(rf, rTp);
  // 6. attention (AC + rel-shifted BD + softmax + P@V), 266 KB dynamic LDS
  const int smem = (QB * 1024 * 2 + 1024) * (int)sizeof(float);
  (void)hipFuncSetAttribute((const void*)attn_kernel,
                            hipFuncAttributeMaxDynamicSharedMemorySize, smem);
  attn_kernel<<<dim3(T_SEQ / QB, N_H, N_B), 256, smem, stream>>>(
      qrw, qrr, kTp, vp, rTp, rf, lens, attnb);
  // 7. output projection + residual -> d_out f32
  gemm_bf16_kernel<<<dim3(D_MODEL / 128, BT / 64), 256, 0, stream>>>(
      attnb, wout_p, out, x, nullptr, D_MODEL, D_MODEL, D_MODEL);
  // 8. second tuple output: lens
  copy_lens<<<1, 32, 0, stream>>>(lens, (int*)d_out + (size_t)BT * D_MODEL);
}
